// TransformerBlock_35837207118382
// MI455X (gfx1250) — matrix operand order repaired and checked by reading, not compiled, not run on hardware
//
#include <hip/hip_runtime.h>
#include <hip/hip_bf16.h>

// Problem constants (match reference)
#define Bc 4
#define Sc 2048
#define Ec 1024
#define Hc 16
#define DHc 64
#define Fc 4096

typedef __attribute__((ext_vector_type(16))) __bf16 v16bf;
typedef __attribute__((ext_vector_type(8)))  float  v8f;
typedef __attribute__((ext_vector_type(4)))  unsigned uint4v;
typedef __attribute__((ext_vector_type(8)))  unsigned uint8v;

union FragBF { v16bf v; unsigned u[8]; };

__device__ __forceinline__ unsigned short bf16bits(float f) {
  __hip_bfloat16 h = __float2bfloat16(f);
  unsigned short u;
  __builtin_memcpy(&u, &h, sizeof(u));
  return u;
}
__device__ __forceinline__ unsigned pack2(unsigned short lo, unsigned short hi) {
  return (unsigned)lo | ((unsigned)hi << 16);
}
// A-fragment K-pair base for (vgpr v, lane-half h): contiguous k = 16h + 2v (matches B)
__device__ __forceinline__ int kApair(int vv, int hh) {
  return 16 * hh + 2 * vv;
}
__device__ __forceinline__ v8f wmma_bf16(const FragBF& a, const FragBF& b, v8f c) {
  return __builtin_amdgcn_wmma_f32_16x16x32_bf16(
      /*neg_a=*/false, a.v, /*neg_b=*/false, b.v,
      /*c_mod=*/(short)0, c, /*reuse_a=*/false, /*reuse_b=*/false);
}

// gfx1250 async LDS-DMA: 16B per lane, tracked by ASYNCcnt
__device__ __forceinline__ void async_copy_b128(unsigned lds_off, const void* gaddr) {
  asm volatile("global_load_async_to_lds_b128 %0, %1, off"
               :: "v"(lds_off), "v"((unsigned long long)gaddr) : "memory");
}
__device__ __forceinline__ void wait_asynccnt0() {
  asm volatile("s_wait_asynccnt 0x0" ::: "memory");
}
__device__ __forceinline__ unsigned lds_offset_of(const void* p) {
  return (unsigned)(unsigned long long)p;  // LDS aperture: low 32 bits = offset
}

// ---------------------------------------------------------------------------
// Tensor Data Mover: load a [128 rows x 32 cols] bf16 tile (row stride = K
// elements) from global into LDS with a single wave-level instruction.
// D# built per CDNA5 ISA ch.8 (group0: count/lds/global/type, group1: dims).
// ---------------------------------------------------------------------------
__device__ __forceinline__ void tdm_load_tile_128x32_bf16(
    unsigned lds_off, const void* gaddr, int K_elems) {
  unsigned long long ga = (unsigned long long)gaddr;
  uint4v g0;
  g0.x = 1u;                                     // count=1, user mode, no gather
  g0.y = lds_off;                                // lds_addr [63:32]
  g0.z = (unsigned)ga;                           // global_addr[31:0]
  g0.w = (unsigned)(ga >> 32) | (2u << 30);      // global_addr[56:32] | type=2
  unsigned K = (unsigned)K_elems;
  uint8v g1;
  g1.s0 = 0x00010000u;                           // wg_mask=0, data_size=1 (2B)
  g1.s1 = (K & 0xFFFFu) << 16;                   // tensor_dim0 lo16 (bits 79:48)
  g1.s2 = (K >> 16) | (128u << 16);              // tensor_dim0 hi16 | tensor_dim1=128 lo16
  g1.s3 = (32u << 16);                           // tensor_dim1 hi16=0 | tile_dim0=32
  g1.s4 = 128u;                                  // tile_dim1=128 | tile_dim2=0
  g1.s5 = K;                                     // tensor_dim0_stride lo32
  g1.s6 = 0u;                                    // stride hi16 | dim1_stride lo16
  g1.s7 = 0u;
  uint4v gz = (uint4v){0u, 0u, 0u, 0u};          // groups 2/3: dims 2..4 unused
  asm volatile("tensor_load_to_lds %0, %1, %2, %3"
               :: "s"(g0), "s"(g1), "s"(gz), "s"(gz) : "memory");
}

// ---------------------------------------------------------------------------
// One-time converts
// ---------------------------------------------------------------------------
__global__ __launch_bounds__(256) void cvt_f32_bf16(
    const float* __restrict__ in, unsigned short* __restrict__ outp, long long n) {
  long long i = ((long long)blockIdx.x * 256 + threadIdx.x) * 4;
  if (i >= n) return;
  float4 v = *(const float4*)(in + i);
  uint2 o;
  o.x = pack2(bf16bits(v.x), bf16bits(v.y));
  o.y = pack2(bf16bits(v.z), bf16bits(v.w));
  *(uint2*)(outp + i) = o;
}

// W[K][N] f32 -> Wt[N][K] bf16 (32x32 LDS tiles)
__global__ __launch_bounds__(256) void transpose_f32_bf16(
    const float* __restrict__ W, unsigned short* __restrict__ Wt, int K, int N) {
  __shared__ unsigned short lt[32][33];
  const int k0 = blockIdx.x * 32, n0 = blockIdx.y * 32;
  const int t = threadIdx.x;
  {
    int kr = t >> 3, nc = (t & 7) * 4;
    const float4 v = *(const float4*)(W + (size_t)(k0 + kr) * N + n0 + nc);
    lt[nc + 0][kr] = bf16bits(v.x);
    lt[nc + 1][kr] = bf16bits(v.y);
    lt[nc + 2][kr] = bf16bits(v.z);
    lt[nc + 3][kr] = bf16bits(v.w);
  }
  __syncthreads();
  {
    int nr = t >> 3, kc = (t & 7) * 4;
    uint2 o;
    o.x = pack2(lt[nr][kc + 0], lt[nr][kc + 1]);
    o.y = pack2(lt[nr][kc + 2], lt[nr][kc + 3]);
    *(uint2*)(Wt + (size_t)(n0 + nr) * K + k0 + kc) = o;
  }
}

// ---------------------------------------------------------------------------
// GEMM: C[M,N] = A[M,K] * Bt[N,K]^T   (bf16 in, bf16 WMMA, f32 or bf16 out)
// 256 threads (8 waves), 128x128 tile, K-step 32, double-buffered LDS.
// A-tile: per-lane global_load_async_to_lds_b128 (ASYNCcnt).
// B-tile (weights): one TDM tensor_load_to_lds per stage (TENSORcnt).
// K-loop unrolled by 2 -> static buffer indices, branch-free inner loop.
// ---------------------------------------------------------------------------
__global__ __launch_bounds__(256) void gemm_wmma_bf16(
    const unsigned short* __restrict__ A,   // [M][K] bf16
    const unsigned short* __restrict__ Bt,  // [N][K] bf16 (pre-transposed)
    void* __restrict__ Cout, int M, int N, int K,
    const float* __restrict__ bias, int relu, int out_bf16) {
  __shared__ __align__(16) unsigned short lA[2][128][32];
  __shared__ __align__(16) unsigned short lB[2][128][32];

  const int tid  = threadIdx.x;
  const int wave = tid >> 5;
  const int lane = tid & 31;
  const int h    = lane >> 4;
  const int l16  = lane & 15;

  const int bm = blockIdx.x * 128;
  const int bn = blockIdx.y * 128;
  const int wm = (wave & 3) * 32;
  const int wn = (wave >> 2) * 64;

  v8f acc[2][4];
#pragma unroll
  for (int mi = 0; mi < 2; ++mi)
#pragma unroll
    for (int ni = 0; ni < 4; ++ni)
#pragma unroll
      for (int e = 0; e < 8; ++e) acc[mi][ni][e] = 0.0f;

  auto stage = [&](int buf, int k0) {
#pragma unroll
    for (int it = 0; it < 2; ++it) {
      int slot = tid + it * 256;        // 0..511 -> (row, 8-elem chunk)
      int r = slot >> 2;
      int c = (slot & 3) * 8;
      async_copy_b128(lds_offset_of(&lA[buf][r][c]), A + (size_t)(bm + r) * K + k0 + c);
    }
    if (wave == 0)
      tdm_load_tile_128x32_bf16(lds_offset_of(&lB[buf][0][0]),
                                Bt + (size_t)bn * K + k0, K);
  };
  auto waitcopies = [&]() {
    wait_asynccnt0();
    if (wave == 0) __builtin_amdgcn_s_wait_tensorcnt((short)0);
    __syncthreads();
  };
  auto compute = [&](int buf) {
    FragBF af[2], bfg[4];
#pragma unroll
    for (int mi = 0; mi < 2; ++mi)
#pragma unroll
      for (int vv = 0; vv < 8; ++vv)
        af[mi].u[vv] = *(const unsigned*)&lA[buf][wm + mi * 16 + l16][kApair(vv, h)];
#pragma unroll
    for (int ni = 0; ni < 4; ++ni)
#pragma unroll
      for (int vv = 0; vv < 8; ++vv)
        bfg[ni].u[vv] = *(const unsigned*)&lB[buf][wn + ni * 16 + l16][16 * h + 2 * vv];
#pragma unroll
    for (int mi = 0; mi < 2; ++mi)
#pragma unroll
      for (int ni = 0; ni < 4; ++ni)
        acc[mi][ni] = wmma_bf16(af[mi], bfg[ni], acc[mi][ni]);
  };

  stage(0, 0);
  waitcopies();
  for (int k0 = 0; k0 < K; k0 += 64) {   // K is a multiple of 64 for all GEMMs
    stage(1, k0 + 32);                   // always in range (k0 <= K-64)
    compute(0);
    waitcopies();
    stage(0, (k0 + 64 < K) ? (k0 + 64) : 0);  // last prefetch wraps (discarded)
    compute(1);
    waitcopies();
  }

  // epilogue: C layout lane(h, n=l16), vgpr r -> row r+8h
#pragma unroll
  for (int mi = 0; mi < 2; ++mi) {
#pragma unroll
    for (int ni = 0; ni < 4; ++ni) {
      int col = bn + wn + ni * 16 + l16;
      float bv = bias ? bias[col] : 0.0f;
#pragma unroll
      for (int r = 0; r < 8; ++r) {
        int row = bm + wm + mi * 16 + r + 8 * h;
        float val = acc[mi][ni][r] + bv;
        if (relu) val = fmaxf(val, 0.0f);
        if (out_bf16)
          ((unsigned short*)Cout)[(size_t)row * N + col] = bf16bits(val);
        else
          ((float*)Cout)[(size_t)row * N + col] = val;
      }
    }
  }
}

// ---------------------------------------------------------------------------
// Flash-style causal MHA on bf16 kqv buffer [B,S,H,3(K,Q,V),DH].
// Grid: (S/128, H, B). 8 waves; wave owns 16 query rows. Output bf16 concat.
// ---------------------------------------------------------------------------
__global__ __launch_bounds__(256) void attn_flash_wmma(
    const unsigned short* __restrict__ kqv, unsigned short* __restrict__ concat) {
  __shared__ unsigned short lP[8][16][32];  // per-wave P staging (bf16)

  const int wave = threadIdx.x >> 5;
  const int lane = threadIdx.x & 31;
  const int h    = lane >> 4;
  const int l16  = lane & 15;

  const int head = blockIdx.y;
  const int b    = blockIdx.z;
  const int q0   = blockIdx.x * 128 + wave * 16;
  const size_t base = (size_t)b * Sc * (3 * Ec);

  FragBF aq[2];
  {
    const unsigned short* qptr = kqv + base + (size_t)(q0 + l16) * (3 * Ec) + (head * 3 + 1) * DHc;
#pragma unroll
    for (int dc = 0; dc < 2; ++dc)
#pragma unroll
      for (int vv = 0; vv < 8; ++vv)
        aq[dc].u[vv] = *(const unsigned*)(qptr + dc * 32 + kApair(vv, h));
  }

  float m_run[8], l_run[8];
  v8f acc[4];
#pragma unroll
  for (int r = 0; r < 8; ++r) { m_run[r] = -__builtin_inff(); l_run[r] = 0.0f; }
#pragma unroll
  for (int dt = 0; dt < 4; ++dt)
#pragma unroll
    for (int e = 0; e < 8; ++e) acc[dt][e] = 0.0f;

  const int qmax = q0 + 15;
  for (int key0 = 0; key0 <= qmax; key0 += 32) {
    FragBF bK[2][2];
#pragma unroll
    for (int nt = 0; nt < 2; ++nt) {
      const unsigned short* kptr = kqv + base + (size_t)(key0 + nt * 16 + l16) * (3 * Ec) + (head * 3 + 0) * DHc;
#pragma unroll
      for (int dc = 0; dc < 2; ++dc)
#pragma unroll
        for (int vv = 0; vv < 8; ++vv)
          bK[nt][dc].u[vv] = *(const unsigned*)(kptr + dc * 32 + 16 * h + 2 * vv);
    }
    v8f s[2];
#pragma unroll
    for (int nt = 0; nt < 2; ++nt) {
#pragma unroll
      for (int e = 0; e < 8; ++e) s[nt][e] = 0.0f;
#pragma unroll
      for (int dc = 0; dc < 2; ++dc) s[nt] = wmma_bf16(aq[dc], bK[nt][dc], s[nt]);
    }
    // scale 1/sqrt(DH) + causal mask
#pragma unroll
    for (int nt = 0; nt < 2; ++nt) {
      int key = key0 + nt * 16 + l16;
#pragma unroll
      for (int r = 0; r < 8; ++r)
        s[nt][r] = (key > q0 + r + 8 * h) ? -__builtin_inff() : s[nt][r] * 0.125f;
    }
    // online softmax: row reductions across the 16 lanes of each half
    float mnew[8];
#pragma unroll
    for (int r = 0; r < 8; ++r) {
      float v = fmaxf(s[0][r], s[1][r]);
#pragma unroll
      for (int mx = 1; mx < 16; mx <<= 1) v = fmaxf(v, __shfl_xor(v, mx, 32));
      mnew[r] = fmaxf(m_run[r], v);
    }
#pragma unroll
    for (int nt = 0; nt < 2; ++nt)
#pragma unroll
      for (int r = 0; r < 8; ++r) s[nt][r] = __expf(s[nt][r] - mnew[r]);
#pragma unroll
    for (int r = 0; r < 8; ++r) {
      float sm = s[0][r] + s[1][r];
#pragma unroll
      for (int mx = 1; mx < 16; mx <<= 1) sm += __shfl_xor(sm, mx, 32);
      float alpha = __expf(m_run[r] - mnew[r]);
      l_run[r] = l_run[r] * alpha + sm;
      m_run[r] = mnew[r];
#pragma unroll
      for (int dt = 0; dt < 4; ++dt) acc[dt][r] *= alpha;
    }
    // C-layout -> A-layout via per-wave LDS
#pragma unroll
    for (int nt = 0; nt < 2; ++nt)
#pragma unroll
      for (int r = 0; r < 8; ++r)
        lP[wave][r + 8 * h][nt * 16 + l16] = bf16bits(s[nt][r]);
    asm volatile("s_wait_dscnt 0x0" ::: "memory");

    FragBF ap;
#pragma unroll
    for (int vv = 0; vv < 8; ++vv)
      ap.u[vv] = *(const unsigned*)&lP[wave][l16][kApair(vv, h)];

    // O += P V
#pragma unroll
    for (int dt = 0; dt < 4; ++dt) {
      FragBF bV;
      const unsigned short* vptr = kqv + base + (size_t)key0 * (3 * Ec) + (head * 3 + 2) * DHc + dt * 16 + l16;
#pragma unroll
      for (int vv = 0; vv < 8; ++vv) {
        int k2 = 16 * h + 2 * vv;
        bV.u[vv] = pack2(vptr[(size_t)k2 * (3 * Ec)], vptr[(size_t)(k2 + 1) * (3 * Ec)]);
      }
      acc[dt] = wmma_bf16(ap, bV, acc[dt]);
    }
  }

#pragma unroll
  for (int dt = 0; dt < 4; ++dt)
#pragma unroll
    for (int r = 0; r < 8; ++r) {
      int q = q0 + r + 8 * h;
      concat[((size_t)b * Sc + q) * Ec + head * DHc + dt * 16 + l16] =
          bf16bits(acc[dt][r] / l_run[r]);
    }
}

// ---------------------------------------------------------------------------
// outf = LayerNorm(x)*g + bb + res ; optionally also bf16 copy for next GEMM
// ---------------------------------------------------------------------------
__global__ __launch_bounds__(256) void ln_residual(
    const float* __restrict__ x, const float* __restrict__ g,
    const float* __restrict__ bb, const float* __restrict__ res,
    float* __restrict__ outf, unsigned short* __restrict__ outb) {
  const int row = blockIdx.x;
  const float* xr = x + (size_t)row * Ec;
  const float* rr = res + (size_t)row * Ec;
  float* orow = outf + (size_t)row * Ec;

  __shared__ float redS[8], redQ[8];
  __shared__ float mu_s, rs_s;

  float sum = 0.0f, sq = 0.0f;
  for (int i = threadIdx.x; i < Ec; i += 256) {
    float v = xr[i];
    sum += v; sq += v * v;
  }
#pragma unroll
  for (int mx = 1; mx < 32; mx <<= 1) {
    sum += __shfl_xor(sum, mx, 32);
    sq  += __shfl_xor(sq,  mx, 32);
  }
  if ((threadIdx.x & 31) == 0) { redS[threadIdx.x >> 5] = sum; redQ[threadIdx.x >> 5] = sq; }
  __syncthreads();
  if (threadIdx.x == 0) {
    float S = 0.0f, Q = 0.0f;
#pragma unroll
    for (int w = 0; w < 8; ++w) { S += redS[w]; Q += redQ[w]; }
    float mu = S * (1.0f / Ec);
    float var = Q * (1.0f / Ec) - mu * mu;
    mu_s = mu;
    rs_s = rsqrtf(var + 1e-6f);
  }
  __syncthreads();
  float mu = mu_s, rs = rs_s;
  for (int i = threadIdx.x; i < Ec; i += 256) {
    float v = (xr[i] - mu) * rs * g[i] + bb[i] + rr[i];
    orow[i] = v;
    if (outb) outb[(size_t)row * Ec + i] = bf16bits(v);
  }
}

// ---------------------------------------------------------------------------
extern "C" void kernel_launch(void* const* d_in, const int* in_sizes, int n_in,
                              void* d_out, int out_size, void* d_ws, size_t ws_size,
                              hipStream_t stream) {
  const float* x      = (const float*)d_in[0];   // [B,S,E]
  const float* KQV    = (const float*)d_in[1];   // [E,3E]
  const float* WO     = (const float*)d_in[2];   // [E,E]
  const float* W_up   = (const float*)d_in[3];   // [E,F]
  const float* b_up   = (const float*)d_in[4];   // [F]
  const float* W_down = (const float*)d_in[5];   // [F,E]
  const float* b_down = (const float*)d_in[6];   // [E]
  const float* ln1_g  = (const float*)d_in[7];
  const float* ln1_b  = (const float*)d_in[8];
  const float* ln2_g  = (const float*)d_in[9];
  const float* ln2_b  = (const float*)d_in[10];
  float* out = (float*)d_out;

  const size_t M = (size_t)Bc * Sc;  // 8192 rows

  char* wsb = (char*)d_ws;
  size_t off = 0;
  auto alloc = [&](size_t bytes) {
    off = (off + 255) & ~(size_t)255;
    void* p = wsb + off;
    off += bytes;
    return p;
  };
  unsigned short* xbf     = (unsigned short*)alloc(M * Ec * 2);
  unsigned short* kqvt    = (unsigned short*)alloc((size_t)3 * Ec * Ec * 2);  // [3E][E]
  unsigned short* wot     = (unsigned short*)alloc((size_t)Ec * Ec * 2);      // [E][E]
  unsigned short* wupt    = (unsigned short*)alloc((size_t)Fc * Ec * 2);      // [F][E]
  unsigned short* wdownt  = (unsigned short*)alloc((size_t)Ec * Fc * 2);      // [E][F]
  unsigned short* kqvbf   = (unsigned short*)alloc(M * 3 * Ec * 2);
  unsigned short* concatb = (unsigned short*)alloc(M * Ec * 2);
  unsigned short* x1bf    = (unsigned short*)alloc(M * Ec * 2);
  unsigned short* hbf     = (unsigned short*)alloc(M * Fc * 2);
  float* attnf = (float*)alloc(M * Ec * 4);
  float* x1f   = (float*)alloc(M * Ec * 4);
  float* fff   = (float*)alloc(M * Ec * 4);

  dim3 blk(256);

  // one-time bf16 conversions / weight transposes
  cvt_f32_bf16<<<dim3((unsigned)(M * Ec / 1024)), blk, 0, stream>>>(x, xbf, (long long)M * Ec);
  transpose_f32_bf16<<<dim3(Ec / 32, 3 * Ec / 32), blk, 0, stream>>>(KQV,    kqvt,   Ec, 3 * Ec);
  transpose_f32_bf16<<<dim3(Ec / 32, Ec / 32),     blk, 0, stream>>>(WO,     wot,    Ec, Ec);
  transpose_f32_bf16<<<dim3(Ec / 32, Fc / 32),     blk, 0, stream>>>(W_up,   wupt,   Ec, Fc);
  transpose_f32_bf16<<<dim3(Fc / 32, Ec / 32),     blk, 0, stream>>>(W_down, wdownt, Fc, Ec);

  // 1) kqv = x @ KQV -> bf16
  gemm_wmma_bf16<<<dim3(M / 128, (3 * Ec) / 128), blk, 0, stream>>>(
      xbf, kqvt, kqvbf, (int)M, 3 * Ec, Ec, nullptr, 0, 1);

  // 2) causal MHA -> concat (bf16)
  attn_flash_wmma<<<dim3(Sc / 128, Hc, Bc), blk, 0, stream>>>(kqvbf, concatb);

  // 3) attn = concat @ WO -> f32
  gemm_wmma_bf16<<<dim3(M / 128, Ec / 128), blk, 0, stream>>>(
      concatb, wot, attnf, (int)M, Ec, Ec, nullptr, 0, 0);

  // 4) x1 = LN1(attn) + x   (f32 + bf16 copies)
  ln_residual<<<dim3((unsigned)M), blk, 0, stream>>>(attnf, ln1_g, ln1_b, x, x1f, x1bf);

  // 5) h = relu(x1 @ W_up + b_up) -> bf16
  gemm_wmma_bf16<<<dim3(M / 128, Fc / 128), blk, 0, stream>>>(
      x1bf, wupt, hbf, (int)M, Fc, Ec, b_up, 1, 1);

  // 6) ff = h @ W_down + b_down -> f32
  gemm_wmma_bf16<<<dim3(M / 128, Ec / 128), blk, 0, stream>>>(
      hbf, wdownt, fff, (int)M, Ec, Fc, b_down, 0, 0);

  // 7) out = LN2(ff) + x1
  ln_residual<<<dim3((unsigned)M), blk, 0, stream>>>(fff, ln2_g, ln2_b, x1f, out, nullptr);
}
